// MatrixNet_13202729468081
// MI455X (gfx1250) — compile-verified, matrix operand order checked by reading, not run on hardware
//
#include <hip/hip_runtime.h>

// ---------------------------------------------------------------------------
// Types for CDNA5 WMMA (wave32, v_wmma_f32_16x16x32_bf16)
// ---------------------------------------------------------------------------
typedef __attribute__((ext_vector_type(8)))  __bf16 bf16x8;
typedef __attribute__((ext_vector_type(16))) __bf16 bf16x16;
typedef __attribute__((ext_vector_type(8)))  float  v8f;

__device__ __forceinline__ float bf2f(unsigned short u) {
  union { unsigned int i; float f; } v; v.i = ((unsigned int)u) << 16; return v.f;
}
__device__ __forceinline__ unsigned short f2bf(float f) {
  union { float f; unsigned int i; } v; v.f = f;
  unsigned int x = v.i;
  unsigned int r = (x + 0x7FFFu + ((x >> 16) & 1u)) >> 16;  // round-nearest-even
  return (unsigned short)r;
}
__device__ __forceinline__ float clip10(float x) {
  return fminf(10.0f, fmaxf(-10.0f, x));
}
// Branchless t^n for integer 0 <= n < 32 (square-and-multiply, selects only).
// For integer deg:  where(deg even, 1, sign(t)) * |t|^deg == t^deg.
__device__ __forceinline__ float ipowf(float t, int n) {
  float p = 1.0f, b = t;
#pragma unroll
  for (int i = 0; i < 5; ++i) {
    p = (n & 1) ? p * b : p;
    b *= b;
    n >>= 1;
  }
  return p;
}

// ---------------------------------------------------------------------------
// Prep: row index of each edge (in_mask has one 1 per column) and deg[e]
// (column sums of e2o_mask).
// ---------------------------------------------------------------------------
__global__ void bp_edge_prep(const float* __restrict__ in_mask,
                             const float* __restrict__ e2o,
                             int* __restrict__ rowidx,
                             float* __restrict__ deg,
                             int K, int E) {
  const int e = blockIdx.x * blockDim.x + threadIdx.x;
  if (e >= E) return;
  int ri = 0;
  for (int k = 0; k < K; ++k) {
    if (in_mask[(size_t)k * E + e] > 0.5f) { ri = k; break; }
  }
  float d = 0.0f;
  for (int r = 0; r < E; ++r) d += e2o[(size_t)r * E + e];
  rowidx[e] = ri;
  deg[e] = d;
}

// ---------------------------------------------------------------------------
// Prep: WT[c][r] = bf16(mask[r][c] * w[r][c])   (mask,w are R x C row-major)
// 32x32 LDS tile transpose, block (32,8).
// ---------------------------------------------------------------------------
__global__ void bp_mask_transpose(const float* __restrict__ m,
                                  const float* __restrict__ w,
                                  unsigned short* __restrict__ outT,
                                  int R, int C) {
  __shared__ float tile[32][33];
  const int x = threadIdx.x, y = threadIdx.y;
  const int cb = blockIdx.x * 32, rb = blockIdx.y * 32;
#pragma unroll
  for (int i = 0; i < 4; ++i) {
    const int r = rb + y + i * 8;
    const size_t idx = (size_t)r * C + cb + x;
    tile[y + i * 8][x] = m[idx] * w[idx];
  }
  __syncthreads();
#pragma unroll
  for (int i = 0; i < 4; ++i) {
    const int c = cb + y + i * 8;
    outT[(size_t)c * R + rb + x] = f2bf(tile[x][y + i * 8]);
  }
}

// ---------------------------------------------------------------------------
// Prep: lv[b][e] = bf16( x[b][rowidx[e]] )
// ---------------------------------------------------------------------------
__global__ void bp_compute_lv(const float* __restrict__ x,
                              const int* __restrict__ rowidx,
                              unsigned short* __restrict__ lvh,
                              int K, int E) {
  const long long idx = (long long)blockIdx.x * blockDim.x + threadIdx.x;
  const int e = (int)(idx % E);
  const int b = (int)(idx / E);
  lvh[idx] = f2bf(x[(size_t)b * K + rowidx[e]]);
}

// ---------------------------------------------------------------------------
// Fused bf16 WMMA GEMM:  C = A[MxK] * BT[NxK]^T   with epilogue
//   EPMODE 0: odd + even(clip)    -> bf16 out
//   EPMODE 1: odd + even(noclip)  -> bf16 out
//   EPMODE 2: odd only (tanh)     -> bf16 out
//   EPMODE 3: plain               -> f32 out
// Block tile 128x128x32; 8 waves (4 along M x 2 along N); wave tile 32x64.
// Double-buffered LDS, padded rows; global_prefetch for tile kt+2.
// Epilogue (modes 0-2) stages the 128x128 tile through LDS so all global
// traffic is coalesced 16B b128 accesses.
// ---------------------------------------------------------------------------
template <int EPMODE>
__global__ __launch_bounds__(256) void bp_gemm_bf16(
    const unsigned short* __restrict__ A,    // [M][K] bf16
    const unsigned short* __restrict__ BT,   // [Ncols][K] bf16 (transposed B)
    void* __restrict__ Cout,                 // bf16 or f32, [M][Ncols]
    const unsigned short* __restrict__ lvh,  // [M][Ncols] bf16 (iv term)
    const float* __restrict__ wv,            // [Ncols]
    const float* __restrict__ deg,           // [Ncols]
    int Ncols, int Kdim) {
  constexpr int BK   = 32;
  constexpr int LDT  = 40;    // padded K-loop LDS row stride (bf16 elems, 80B)
  constexpr int TBUF = 5120;  // 128*LDT
  // 40KB shared: [A buf0][A buf1][B buf0][B buf1]; reused as epilogue stage.
  __shared__ __align__(16) unsigned short smem[4 * TBUF];
  unsigned short* const As = smem;            // 2 buffers of TBUF
  unsigned short* const Bs = smem + 2 * TBUF; // 2 buffers of TBUF

  const int tid   = threadIdx.x;
  const int lane  = tid & 31;
  const int wave  = tid >> 5;
  const int waveM = wave & 3;   // 4 waves along M
  const int waveN = wave >> 2;  // 2 waves along N
  const int half  = lane >> 4;  // ISA 16-bit A/B lane split
  const int l15   = lane & 15;

  const int bm = blockIdx.y * 128;
  const int bn = blockIdx.x * 128;

  // global->LDS staging: each thread moves 2x16B for A and 2x16B for B
  const int r0 = tid >> 2;
  const int c0 = (tid & 3) * 8;
  const unsigned short* gA0 = A  + (size_t)(bm + r0)      * Kdim + c0;
  const unsigned short* gA1 = A  + (size_t)(bm + r0 + 64) * Kdim + c0;
  const unsigned short* gB0 = BT + (size_t)(bn + r0)      * Kdim + c0;
  const unsigned short* gB1 = BT + (size_t)(bn + r0 + 64) * Kdim + c0;

  v8f acc[2][4];
#pragma unroll
  for (int mt = 0; mt < 2; ++mt)
#pragma unroll
    for (int nt = 0; nt < 4; ++nt)
      acc[mt][nt] = (v8f){0.f, 0.f, 0.f, 0.f, 0.f, 0.f, 0.f, 0.f};

  const int kTiles = Kdim / BK;

  // prologue: stage tile 0
  uint4 fa0 = *(const uint4*)gA0;
  uint4 fa1 = *(const uint4*)gA1;
  uint4 fb0 = *(const uint4*)gB0;
  uint4 fb1 = *(const uint4*)gB1;
  *(uint4*)&As[r0 * LDT + c0]        = fa0;
  *(uint4*)&As[(r0 + 64) * LDT + c0] = fa1;
  *(uint4*)&Bs[r0 * LDT + c0]        = fb0;
  *(uint4*)&Bs[(r0 + 64) * LDT + c0] = fb1;
  __syncthreads();

  int cur = 0;
  for (int kt = 0; kt < kTiles; ++kt) {
    const bool more = (kt + 1) < kTiles;
    if (more) {  // issue global loads for next tile; latency hidden by WMMAs
      const size_t ko = (size_t)(kt + 1) * BK;
      fa0 = *(const uint4*)(gA0 + ko);
      fa1 = *(const uint4*)(gA1 + ko);
      fb0 = *(const uint4*)(gB0 + ko);
      fb1 = *(const uint4*)(gB1 + ko);
      if (kt + 2 < kTiles) {
        __builtin_prefetch(gA0 + ko + BK, 0, 0);  // global_prefetch_b8
        __builtin_prefetch(gB0 + ko + BK, 0, 0);
      }
    }

    // Fragments per the CDNA5 16-bit operand layout:
    // lanes 0-15: K 0..7 | 16..23 ; lanes 16-31: K 8..15 | 24..31
    bf16x16 aF[2], bF[4];
#pragma unroll
    for (int mt = 0; mt < 2; ++mt) {
      const int r = waveM * 32 + mt * 16 + l15;
      const unsigned short* p = &As[cur * TBUF + r * LDT + half * 8];
      bf16x8 lo = *(const bf16x8*)p;
      bf16x8 hi = *(const bf16x8*)(p + 16);
      aF[mt] = __builtin_shufflevector(lo, hi, 0, 1, 2, 3, 4, 5, 6, 7, 8, 9, 10,
                                       11, 12, 13, 14, 15);
    }
#pragma unroll
    for (int nt = 0; nt < 4; ++nt) {
      const int r = waveN * 64 + nt * 16 + l15;
      const unsigned short* p = &Bs[cur * TBUF + r * LDT + half * 8];
      bf16x8 lo = *(const bf16x8*)p;
      bf16x8 hi = *(const bf16x8*)(p + 16);
      bF[nt] = __builtin_shufflevector(lo, hi, 0, 1, 2, 3, 4, 5, 6, 7, 8, 9, 10,
                                       11, 12, 13, 14, 15);
    }

#pragma unroll
    for (int mt = 0; mt < 2; ++mt)
#pragma unroll
      for (int nt = 0; nt < 4; ++nt)
        acc[mt][nt] = __builtin_amdgcn_wmma_f32_16x16x32_bf16(
            false, aF[mt], false, bF[nt], (short)0, acc[mt][nt], false, false);

    if (more) {  // write next tile into the other buffer
      const int nxt = cur ^ 1;
      *(uint4*)&As[nxt * TBUF + r0 * LDT + c0]        = fa0;
      *(uint4*)&As[nxt * TBUF + (r0 + 64) * LDT + c0] = fa1;
      *(uint4*)&Bs[nxt * TBUF + r0 * LDT + c0]        = fb0;
      *(uint4*)&Bs[nxt * TBUF + (r0 + 64) * LDT + c0] = fb1;
    }
    __syncthreads();
    cur ^= 1;
  }

  // ---------------------------------------------------------------------
  // Epilogue. C/D layout: VGPR v, lanes 0-15 -> M=v,N=lane; 16-31 -> M=8+v.
  // ---------------------------------------------------------------------
  if (EPMODE != 3) {
    // LDS buffers are dead now; reuse as a 128 x 136 bf16 staging tile
    // (stride 136 keeps rows 16B-aligned for b128 LDS ops).
    constexpr int SLD = 136;
    unsigned short* const stage = smem;

    // 1) coalesced load of the lv tile into LDS (16B per thread x 8)
    {
      const unsigned short* gLV = lvh + (size_t)bm * Ncols + bn;
#pragma unroll
      for (int i = 0; i < 8; ++i) {
        const int g   = tid + i * 256;
        const int row = g >> 4;
        const int c8  = (g & 15) * 8;
        *(uint4*)&stage[row * SLD + c8] =
            *(const uint4*)(gLV + (size_t)row * Ncols + c8);
      }
    }
    __syncthreads();

    // 2) per-element transform, in place (same lane reads+writes same addr)
#pragma unroll
    for (int nt = 0; nt < 4; ++nt) {
      const int colL = waveN * 64 + nt * 16 + l15;
      const int nG   = bn + colL;
      const float wvn = wv[nG];
      int dn = 0;
      if (EPMODE != 2) dn = (int)(deg[nG] + 0.5f);
#pragma unroll
      for (int mt = 0; mt < 2; ++mt) {
        const int rowL0 = waveM * 32 + mt * 16 + half * 8;
        v8f c = acc[mt][nt];
#pragma unroll
        for (int v = 0; v < 8; ++v) {
          unsigned short* sp = &stage[(rowL0 + v) * SLD + colL];
          float o = c[v] + bf2f(*sp) * wvn;
          o = clip10(o);
          // tanh(o/2) = (e^o - 1)/(e^o + 1); o in [-10,10] -> no overflow
          const float ex = __expf(o);
          const float t  = (ex - 1.0f) / (ex + 1.0f);
          float outv;
          if (EPMODE == 2) {
            outv = t;
          } else {
            float p = ipowf(t, dn);  // sign-handled |t|^deg for integer deg
            if (EPMODE == 0) p = clip10(p);
            outv = __logf((1.0f + p) / (1.0f - p));
          }
          *sp = f2bf(outv);
        }
      }
    }
    __syncthreads();

    // 3) coalesced store of the result tile (16B per thread x 8)
    {
      unsigned short* gO = (unsigned short*)Cout + (size_t)bm * Ncols + bn;
#pragma unroll
      for (int i = 0; i < 8; ++i) {
        const int g   = tid + i * 256;
        const int row = g >> 4;
        const int c8  = (g & 15) * 8;
        *(uint4*)(gO + (size_t)row * Ncols + c8) =
            *(const uint4*)&stage[row * SLD + c8];
      }
    }
  } else {
    // plain f32 store (output projection)
#pragma unroll
    for (int nt = 0; nt < 4; ++nt) {
      const int nG = bn + waveN * 64 + nt * 16 + l15;
#pragma unroll
      for (int mt = 0; mt < 2; ++mt) {
        const int mB = bm + waveM * 32 + mt * 16 + half * 8;
        v8f c = acc[mt][nt];
#pragma unroll
        for (int v = 0; v < 8; ++v)
          ((float*)Cout)[(size_t)(mB + v) * Ncols + nG] = c[v];
      }
    }
  }
}

// ---------------------------------------------------------------------------
// Host launcher
// ---------------------------------------------------------------------------
extern "C" void kernel_launch(void* const* d_in, const int* in_sizes, int n_in,
                              void* d_out, int out_size, void* d_ws,
                              size_t ws_size, hipStream_t stream) {
  (void)in_sizes; (void)n_in; (void)out_size; (void)ws_size;
  constexpr int K = 256, N = 512, E = 2048, BATCH = 8192;

  const float* x        = (const float*)d_in[0];
  const float* in_mask  = (const float*)d_in[1];
  const float* out_mask = (const float*)d_in[2];
  const float* o2e      = (const float*)d_in[3];
  const float* e2o      = (const float*)d_in[4];
  const float* wv1      = (const float*)d_in[5];
  const float* we1      = (const float*)d_in[6];
  const float* wv2      = (const float*)d_in[7];
  const float* we2      = (const float*)d_in[8];
  const float* wv3      = (const float*)d_in[9];
  const float* we3      = (const float*)d_in[10];
  const float* w_out    = (const float*)d_in[11];

  // workspace carve-up (256B aligned)
  char* ws = (char*)d_ws;
  size_t off = 0;
  auto alloc = [&](size_t bytes) -> void* {
    void* p = ws + off;
    off = (off + bytes + 255) & ~(size_t)255;
    return p;
  };
  int*            rowidx = (int*)alloc((size_t)E * 4);
  float*          deg    = (float*)alloc((size_t)E * 4);
  unsigned short* Wt1    = (unsigned short*)alloc((size_t)E * E * 2);
  unsigned short* Wt2    = (unsigned short*)alloc((size_t)E * E * 2);
  unsigned short* Wt3    = (unsigned short*)alloc((size_t)E * E * 2);
  unsigned short* WoutT  = (unsigned short*)alloc((size_t)N * E * 2);
  unsigned short* lvh    = (unsigned short*)alloc((size_t)BATCH * E * 2);
  unsigned short* actA   = (unsigned short*)alloc((size_t)BATCH * E * 2);
  unsigned short* actB   = (unsigned short*)alloc((size_t)BATCH * E * 2);

  // --- prep ---
  bp_edge_prep<<<(E + 255) / 256, 256, 0, stream>>>(in_mask, e2o, rowidx, deg,
                                                    K, E);
  dim3 tb(32, 8);
  bp_mask_transpose<<<dim3(E / 32, E / 32), tb, 0, stream>>>(o2e, we1, Wt1, E, E);
  bp_mask_transpose<<<dim3(E / 32, E / 32), tb, 0, stream>>>(o2e, we2, Wt2, E, E);
  bp_mask_transpose<<<dim3(E / 32, E / 32), tb, 0, stream>>>(o2e, we3, Wt3, E, E);
  bp_mask_transpose<<<dim3(N / 32, E / 32), tb, 0, stream>>>(out_mask, w_out,
                                                             WoutT, E, N);
  bp_compute_lv<<<((size_t)BATCH * E) / 256, 256, 0, stream>>>(x, rowidx, lvh,
                                                               K, E);

  // --- three BP layers + output projection ---
  dim3 gL(E / 128, BATCH / 128);
  bp_gemm_bf16<0><<<gL, 256, 0, stream>>>(lvh, Wt1, actA, lvh, wv1, deg, E, E);
  bp_gemm_bf16<1><<<gL, 256, 0, stream>>>(actA, Wt2, actB, lvh, wv2, deg, E, E);
  bp_gemm_bf16<2><<<gL, 256, 0, stream>>>(actB, Wt3, actA, lvh, wv3, deg, E, E);
  dim3 gO(N / 128, BATCH / 128);
  bp_gemm_bf16<3><<<gO, 256, 0, stream>>>(actA, WoutT, (float*)d_out, nullptr,
                                          nullptr, nullptr, N, E);
}
